// AttentionBlock_56040733278417
// MI455X (gfx1250) — compile-verified, matrix operand order checked by reading, not run on hardware
//
#include <hip/hip_runtime.h>
#include <hip/hip_bf16.h>

// ---------------------------------------------------------------------------
// MI455X (gfx1250, wave32) fused attention block.
// GEMMs use v_wmma_f32_16x16x32_f16 (f16 in, f32 accumulate).
// Attention K/Vt tile staging uses CDNA5 async global->LDS copies
// (global_load_async_to_lds_b128, ASYNCcnt ordered).
//
// Fragment layouts (wave32, per CDNA5 ISA 7.12.2):
//   A (16x32 f16): lane<16 -> row=lane,   halves K=kb..kb+7 and kb+16..kb+23
//                  lane>=16 -> row=lane-16, kb=8
//   B (32x16 f16): mirror of A with rows = N; i.e. source memory must be
//                  [N][K] row-major, K contiguous per lane.
//   C/D (16x16 f32, 8 VGPRs): elem i -> M = i + (lane>=16 ? 8 : 0), N = lane&15
// ---------------------------------------------------------------------------

typedef __attribute__((ext_vector_type(16))) _Float16 v16h;
typedef __attribute__((ext_vector_type(8)))  _Float16 v8h;
typedef __attribute__((ext_vector_type(4)))  _Float16 v4h;
typedef __attribute__((ext_vector_type(8)))  float    v8f;
typedef __attribute__((ext_vector_type(4)))  float    v4f;

union F16x16 { v16h v; v8h h[2]; };

// Load a 16x32 f16 fragment (A-operand, or B-operand from [N][K] row-major).
// ld in halves; row base must be 16B aligned (ld multiple of 8).
__device__ __forceinline__ v16h load_frag16(const _Float16* p, int ld) {
    const int lane = threadIdx.x & 31;
    const int row  = lane & 15;
    const int kb   = (lane >> 4) << 3;          // 0 or 8
    F16x16 f;
    f.h[0] = *(const v8h*)(p + (size_t)row * ld + kb);
    f.h[1] = *(const v8h*)(p + (size_t)row * ld + kb + 16);
    return f.v;
}

// Same fragment pattern but converting from f32 source on the fly.
__device__ __forceinline__ v16h load_frag16_f32(const float* p, int ld) {
    const int lane = threadIdx.x & 31;
    const int row  = lane & 15;
    const int kb   = (lane >> 4) << 3;
    const float* q = p + (size_t)row * ld + kb;
    v4f a0 = *(const v4f*)(q);
    v4f a1 = *(const v4f*)(q + 4);
    v4f a2 = *(const v4f*)(q + 16);
    v4f a3 = *(const v4f*)(q + 20);
    F16x16 f;
#pragma unroll
    for (int j = 0; j < 4; ++j) {
        f.h[0][j]     = (_Float16)a0[j];
        f.h[0][j + 4] = (_Float16)a1[j];
        f.h[1][j]     = (_Float16)a2[j];
        f.h[1][j + 4] = (_Float16)a3[j];
    }
    return f.v;
}

// CDNA5 async copy: 16B global -> LDS, bypassing VGPRs (ASYNCcnt tracked).
// imm_off is added to BOTH the global and LDS addresses by hardware.
__device__ __forceinline__ void async_copy_b128(const void* gptr, void* lptr) {
    unsigned lds_addr = (unsigned)(uintptr_t)lptr;   // LDS offset = addr[31:0]
    asm volatile("global_load_async_to_lds_b128 %0, %1, off"
                 :: "v"(lds_addr), "v"(gptr) : "memory");
}
__device__ __forceinline__ void async_copy_b128_off16(const void* gptr, void* lptr) {
    unsigned lds_addr = (unsigned)(uintptr_t)lptr;
    asm volatile("global_load_async_to_lds_b128 %0, %1, off offset:16"
                 :: "v"(lds_addr), "v"(gptr) : "memory");
}
__device__ __forceinline__ void wait_async0() {
    asm volatile("s_wait_asynccnt 0" ::: "memory");
}

#define WMMA_F16(A, B, C) \
    __builtin_amdgcn_wmma_f32_16x16x32_f16(false, (A), false, (B), (short)0, (C), false, false)

// ---------------------------------------------------------------------------
// Kernel 1: projection GEMM  out = X @ W.T + bias   (X:[8192,1024] W:[1024,1024])
// transposeOut==0: out f16 as [b, head, s, d]   (Q, K layout)
// transposeOut==1: out f16 as [b, head, d, s]   (V-transposed layout for P@V)
// WG: 256 thr (8 waves), tile 64(M) x 64(N), K steps of 32.
// ---------------------------------------------------------------------------
__global__ __launch_bounds__(256)
void proj_kernel(const float* __restrict__ X, const float* __restrict__ W,
                 const float* __restrict__ bias, _Float16* __restrict__ out,
                 int transposeOut) {
    __shared__ _Float16 Asub[64][40];
    __shared__ _Float16 Bsub[64][40];
    const int m0 = blockIdx.x * 64;
    const int n0 = blockIdx.y * 64;
    const int t = threadIdx.x, lane = t & 31, wave = t >> 5;
    const int wm = wave & 3, wn = wave >> 2;          // 4 x 2 wave grid

    v8f acc0 = {}; v8f acc1 = {};

    for (int k0 = 0; k0 < 1024; k0 += 32) {
        // Stage A and B tiles (f32 -> f16), 64x32 each.
#pragma unroll
        for (int it = 0; it < 2; ++it) {
            int idx = t + it * 256;                   // 0..511
            int row = idx >> 3;                       // 0..63
            int c4  = (idx & 7) << 2;                 // 0..28
            v4f av = *(const v4f*)(X + (size_t)(m0 + row) * 1024 + k0 + c4);
            v4f bv = *(const v4f*)(W + (size_t)(n0 + row) * 1024 + k0 + c4);
            v4h ah, bh;
#pragma unroll
            for (int j = 0; j < 4; ++j) { ah[j] = (_Float16)av[j]; bh[j] = (_Float16)bv[j]; }
            *(v4h*)&Asub[row][c4] = ah;
            *(v4h*)&Bsub[row][c4] = bh;
        }
        __syncthreads();

        v16h af  = load_frag16(&Asub[wm * 16][0], 40);
        v16h bf0 = load_frag16(&Bsub[wn * 32][0], 40);
        v16h bf1 = load_frag16(&Bsub[wn * 32 + 16][0], 40);
        acc0 = WMMA_F16(af, bf0, acc0);
        acc1 = WMMA_F16(af, bf1, acc1);
        __syncthreads();
    }

    // Epilogue: bias + store to head-split layout.
    const int mrow0 = m0 + wm * 16 + ((lane >> 4) << 3);
#pragma unroll
    for (int nt = 0; nt < 2; ++nt) {
        const v8f acc = nt ? acc1 : acc0;
        int n = n0 + wn * 32 + nt * 16 + (lane & 15);
        int head = n >> 6, d = n & 63;
        float bv = bias[n];
        if (transposeOut) {
            // [b, head, d, s]: 8 consecutive s per lane -> one b128 store
            v8h pk;
#pragma unroll
            for (int i = 0; i < 8; ++i) pk[i] = (_Float16)(acc[i] + bv);
            int m = mrow0;                            // 8-aligned, same batch
            int b = m >> 10, s = m & 1023;
            *(v8h*)(out + (((size_t)((b * 16 + head) * 64 + d)) << 10) + s) = pk;
        } else {
#pragma unroll
            for (int i = 0; i < 8; ++i) {
                int m = mrow0 + i;
                int b = m >> 10, s = m & 1023;
                out[(((size_t)((b * 16 + head) * 1024 + s)) << 6) + d] =
                    (_Float16)(acc[i] + bv);
            }
        }
    }
}

// ---------------------------------------------------------------------------
// Kernel 2: flash-style attention.  One WG per (b*16+head, qblock of 128).
// Wave w owns 16 q rows. K/Vt 64x64 tiles staged in LDS (shared by all waves)
// via CDNA5 async global->LDS copies.
// ---------------------------------------------------------------------------
__global__ __launch_bounds__(256)
void attn_kernel(const _Float16* __restrict__ Qh, const _Float16* __restrict__ Kh,
                 const _Float16* __restrict__ Vth, const float* __restrict__ mask,
                 _Float16* __restrict__ ctx) {
    __shared__ _Float16 Ksub[64][80];
    __shared__ _Float16 Vsub[64][80];
    __shared__ _Float16 Psub[8][16][80];

    const int bh = blockIdx.x;                        // 0..127
    const int b  = bh >> 4;
    const int t = threadIdx.x, lane = t & 31, wave = t >> 5;
    const int q0 = blockIdx.y * 128 + wave * 16;

    const _Float16* Qb = Qh  + (size_t)bh * 1024 * 64;
    const _Float16* Kb = Kh  + (size_t)bh * 1024 * 64;
    const _Float16* Vb = Vth + (size_t)bh * 64 * 1024;
    const float*    mk = mask + (size_t)b * 1024;

    const v16h qa0 = load_frag16(Qb + (size_t)q0 * 64, 64);       // d = 0..31
    const v16h qa1 = load_frag16(Qb + (size_t)q0 * 64 + 32, 64);  // d = 32..63

    v8f acc[4] = {};                                  // ctx accum, 4 d-tiles
    float rmax[8], rsum[8];
#pragma unroll
    for (int i = 0; i < 8; ++i) { rmax[i] = -1e30f; rsum[i] = 0.0f; }
    const float scale = 0.125f;                       // 1/sqrt(64)

    for (int kv0 = 0; kv0 < 1024; kv0 += 64) {
        {   // async-stage K (row-major [kv][d]) and Vt (row-major [d][kv]) tiles
            int row = t >> 2;                         // 0..63
            int c8  = (t & 3) << 4;                   // 0,16,32,48
            const _Float16* gk = Kb + (size_t)(kv0 + row) * 64 + c8;
            const _Float16* gv = Vb + (size_t)row * 1024 + kv0 + c8;
            async_copy_b128(gk, &Ksub[row][c8]);
            async_copy_b128_off16(gk, &Ksub[row][c8]);     // halves c8+8..c8+15
            async_copy_b128(gv, &Vsub[row][c8]);
            async_copy_b128_off16(gv, &Vsub[row][c8]);
            wait_async0();
        }
        __syncthreads();

        // scores 16 x 64 (4 kv-subtiles), K-dim = d = 64 -> 2 WMMAs each
        v8f sc[4];
#pragma unroll
        for (int nt = 0; nt < 4; ++nt) {
            v16h kb0 = load_frag16(&Ksub[nt * 16][0], 80);
            v16h kb1 = load_frag16(&Ksub[nt * 16][32], 80);
            v8f c = {};
            c = WMMA_F16(qa0, kb0, c);
            c = WMMA_F16(qa1, kb1, c);
            float mv   = mk[kv0 + nt * 16 + (lane & 15)];
            float madd = (1.0f - mv) * -10000.0f;
#pragma unroll
            for (int i = 0; i < 8; ++i) sc[nt][i] = c[i] * scale + madd;
        }

        // online softmax per row slot (shfl masks <=8 stay in 16-lane half)
#pragma unroll
        for (int i = 0; i < 8; ++i) {
            float tmax = fmaxf(fmaxf(sc[0][i], sc[1][i]), fmaxf(sc[2][i], sc[3][i]));
#pragma unroll
            for (int m = 8; m >= 1; m >>= 1) tmax = fmaxf(tmax, __shfl_xor(tmax, m, 32));
            float nm    = fmaxf(rmax[i], tmax);
            float alpha = __expf(rmax[i] - nm);
            rmax[i] = nm;
            float ps = 0.0f;
#pragma unroll
            for (int nt = 0; nt < 4; ++nt) {
                float p = __expf(sc[nt][i] - nm);
                sc[nt][i] = p;
                ps += p;
            }
#pragma unroll
            for (int m = 8; m >= 1; m >>= 1) ps += __shfl_xor(ps, m, 32);
            rsum[i] = rsum[i] * alpha + ps;
#pragma unroll
            for (int dt = 0; dt < 4; ++dt) acc[dt][i] *= alpha;
        }

        // Transpose P (C-layout) -> row-major [16][64] in per-wave LDS
        {
            int mb = (lane >> 4) << 3;
#pragma unroll
            for (int nt = 0; nt < 4; ++nt)
#pragma unroll
                for (int i = 0; i < 8; ++i)
                    Psub[wave][mb + i][nt * 16 + (lane & 15)] = (_Float16)sc[nt][i];
        }
        // wave-internal LDS RAW: wait for DS stores, no workgroup barrier needed
        asm volatile("s_wait_dscnt 0" ::: "memory");

        v16h pa0 = load_frag16(&Psub[wave][0][0], 80);   // kv 0..31
        v16h pa1 = load_frag16(&Psub[wave][0][32], 80);  // kv 32..63
#pragma unroll
        for (int dt = 0; dt < 4; ++dt) {
            v16h vb0 = load_frag16(&Vsub[dt * 16][0], 80);
            v16h vb1 = load_frag16(&Vsub[dt * 16][32], 80);
            acc[dt] = WMMA_F16(pa0, vb0, acc[dt]);
            acc[dt] = WMMA_F16(pa1, vb1, acc[dt]);
        }
        __syncthreads();
    }

    // finalize: ctx f16 [b, s, H] with H col = head*64 + d
    const int head = bh & 15;
#pragma unroll
    for (int dt = 0; dt < 4; ++dt) {
        int h = head * 64 + dt * 16 + (lane & 15);
#pragma unroll
        for (int i = 0; i < 8; ++i) {
            int q = q0 + ((lane >> 4) << 3) + i;
            ctx[(((size_t)(b * 1024 + q)) << 10) + h] = (_Float16)(acc[dt][i] / rsum[i]);
        }
    }
}

// ---------------------------------------------------------------------------
// Kernel 3: out = LN(ctx @ Wo.T + bo + residual).  One WG per 16 rows; each
// wave owns 128 cols (8 WMMA n-tiles) over full K=1024, so the LayerNorm over
// H=1024 fuses inside the workgroup.
// ---------------------------------------------------------------------------
__global__ __launch_bounds__(256)
void out_kernel(const _Float16* __restrict__ ctx, const float* __restrict__ Wo,
                const float* __restrict__ bo, const float* __restrict__ resid,
                const float* __restrict__ g, const float* __restrict__ beta,
                float* __restrict__ out) {
    __shared__ float Crow[16][1024];                  // 64 KB
    const int r0 = blockIdx.x * 16;
    const int t = threadIdx.x, lane = t & 31, wave = t >> 5;
    const int n0 = wave * 128;

    v8f acc[8] = {};
    for (int k0 = 0; k0 < 1024; k0 += 32) {
        v16h af = load_frag16(ctx + ((size_t)r0 << 10) + k0, 1024);
#pragma unroll
        for (int nt = 0; nt < 8; ++nt) {
            v16h bf = load_frag16_f32(Wo + (size_t)(n0 + nt * 16) * 1024 + k0, 1024);
            acc[nt] = WMMA_F16(af, bf, acc[nt]);
        }
    }

    // bias + park full rows in LDS for the fused LayerNorm
#pragma unroll
    for (int nt = 0; nt < 8; ++nt) {
        int n = n0 + nt * 16 + (lane & 15);
        float bb = bo[n];
#pragma unroll
        for (int i = 0; i < 8; ++i) {
            int m = ((lane >> 4) << 3) + i;
            Crow[m][n] = acc[nt][i] + bb;
        }
    }
    __syncthreads();

    // LayerNorm: 16 threads per row, 64 elems each; 16-lane shuffle reduce
    {
        int row = t >> 4, c0 = (t & 15) * 64;
        const float* rp = resid + ((size_t)(r0 + row) << 10);
        float s = 0.0f, ss = 0.0f;
#pragma unroll 8
        for (int c = 0; c < 64; ++c) {
            float x = Crow[row][c0 + c] + rp[c0 + c];
            s += x; ss += x * x;
        }
#pragma unroll
        for (int m = 8; m >= 1; m >>= 1) {
            s  += __shfl_xor(s,  m, 32);
            ss += __shfl_xor(ss, m, 32);
        }
        float mean = s * (1.0f / 1024.0f);
        float var  = ss * (1.0f / 1024.0f) - mean * mean;
        float rstd = rsqrtf(var + 1e-7f);
        float* op = out + ((size_t)(r0 + row) << 10);
#pragma unroll 8
        for (int c = 0; c < 64; ++c) {
            int cc = c0 + c;
            float x = Crow[row][cc] + rp[cc];
            op[cc] = (x - mean) * rstd * g[cc] + beta[cc];
        }
    }
}

// ---------------------------------------------------------------------------
extern "C" void kernel_launch(void* const* d_in, const int* in_sizes, int n_in,
                              void* d_out, int out_size, void* d_ws, size_t ws_size,
                              hipStream_t stream) {
    const float* q    = (const float*)d_in[0];
    const float* k    = (const float*)d_in[1];
    const float* v    = (const float*)d_in[2];
    const float* mask = (const float*)d_in[3];
    const float* Wq   = (const float*)d_in[4];
    const float* bq   = (const float*)d_in[5];
    const float* Wk   = (const float*)d_in[6];
    const float* bk   = (const float*)d_in[7];
    const float* Wv   = (const float*)d_in[8];
    const float* bv   = (const float*)d_in[9];
    const float* Wo   = (const float*)d_in[10];
    const float* bo   = (const float*)d_in[11];
    const float* ln_g = (const float*)d_in[12];
    const float* ln_b = (const float*)d_in[13];

    const size_t TOK = (size_t)8 * 1024 * 1024;       // B*S*H f16 elements
    _Float16* Qh   = (_Float16*)d_ws;
    _Float16* Kh   = Qh + TOK;
    _Float16* Vth  = Kh + TOK;                        // stored [b,h,d,s]
    _Float16* ctxh = Vth + TOK;

    dim3 gP(8192 / 64, 1024 / 64);                    // 128 x 16
    proj_kernel<<<gP, 256, 0, stream>>>(q, Wq, bq, Qh, 0);
    proj_kernel<<<gP, 256, 0, stream>>>(k, Wk, bk, Kh, 0);
    proj_kernel<<<gP, 256, 0, stream>>>(v, Wv, bv, Vth, 1);

    attn_kernel<<<dim3(128, 8), 256, 0, stream>>>(Qh, Kh, Vth, mask, ctxh);

    out_kernel<<<8192 / 16, 256, 0, stream>>>(ctxh, Wo, bo, q, ln_g, ln_b, (float*)d_out);
}
